// AttentionLayer_55542517072361
// MI455X (gfx1250) — matrix operand order repaired and checked by reading, not compiled, not run on hardware
//
#include <hip/hip_runtime.h>

// ---- CDNA5 (gfx1250, wave32) flash-attention for SAGAN attention layer ----
// Matmuls: v_wmma_f32_16x16x32_f16. V staged into LDS by the Tensor Data Mover
// (tensor_load_to_lds + s_wait_tensorcnt), double-buffered to overlap WMMAs.

typedef __attribute__((ext_vector_type(16))) _Float16 v16h;
typedef __attribute__((ext_vector_type(8)))  _Float16 v8h;
typedef __attribute__((ext_vector_type(8)))  float    v8f;
typedef __attribute__((ext_vector_type(4)))  int      int4v;
typedef __attribute__((ext_vector_type(8)))  int      int8v;
typedef __attribute__((ext_vector_type(4)))  unsigned uint4v;

#define WMMA16(a, b, c) \
  __builtin_amdgcn_wmma_f32_16x16x32_f16(false, (a), false, (b), (short)0, (c), false, false)

#if __has_builtin(__builtin_amdgcn_tensor_load_to_lds) && \
    __has_builtin(__builtin_amdgcn_s_wait_tensorcnt)
#define HAS_TDM 1
#else
#define HAS_TDM 0
#endif

#if __has_builtin(__builtin_amdgcn_global_load_async_to_lds_b128)
#define HAS_ASYNC_LDS 1
#if __has_builtin(__builtin_amdgcn_s_wait_asynccnt)
#define ASYNC_WAIT() __builtin_amdgcn_s_wait_asynccnt(0)
#else
#define ASYNC_WAIT() asm volatile("s_wait_asynccnt 0" ::: "memory")
#endif
#else
#define HAS_ASYNC_LDS 0
#define ASYNC_WAIT() ((void)0)
#endif

static constexpr int kB  = 4;
static constexpr int kN  = 4096;   // W*W
static constexpr int kD  = 64;     // head dim
static constexpr int kC  = 512;    // channels (C == C2)
static constexpr int kKC = 64;     // keys per chunk
static constexpr int kNT = kN / kKC;
static constexpr int kVStride = 72;  // 64 keys + 8 pad (bank spread; == TDM pad 4 DW / 32 DW)
static constexpr int kPStride = 72;

__device__ __forceinline__ v16h cat16(v8h lo, v8h hi) {
  v16h r;
#pragma unroll
  for (int i = 0; i < 8; ++i) { r[i] = lo[i]; r[i + 8] = hi[i]; }
  return r;
}

__device__ __forceinline__ v8f vzero8() {
  v8f z = {0.f, 0.f, 0.f, 0.f, 0.f, 0.f, 0.f, 0.f};
  return z;
}

// -------- transpose + cast: vT[b][c][n] = (f16) h[b][n][c], 64x64 tiles --------
__global__ __launch_bounds__(256) void transpose_cast_kernel(
    const float* __restrict__ src, _Float16* __restrict__ dst) {
  __shared__ __attribute__((aligned(16))) _Float16 tile[64 * kVStride];
  const int tid = threadIdx.x;
  const int b = blockIdx.z;
  const int n0 = blockIdx.x * 64;
  const int c0 = blockIdx.y * 64;
#pragma unroll
  for (int it = 0; it < 4; ++it) {
    int flat = it * 256 + tid;           // 1024 float4
    int n  = flat >> 4;
    int c4 = flat & 15;
    float4 f = *(const float4*)(src + ((size_t)(b * kN + n0 + n)) * kC + c0 + c4 * 4);
    tile[(c4 * 4 + 0) * kVStride + n] = (_Float16)f.x;
    tile[(c4 * 4 + 1) * kVStride + n] = (_Float16)f.y;
    tile[(c4 * 4 + 2) * kVStride + n] = (_Float16)f.z;
    tile[(c4 * 4 + 3) * kVStride + n] = (_Float16)f.w;
  }
  __syncthreads();
#pragma unroll
  for (int it = 0; it < 2; ++it) {
    int flat = it * 256 + tid;           // 512 v8h
    int c  = flat >> 3;
    int k8 = flat & 7;
    v8h val = *(const v8h*)(&tile[c * kVStride + k8 * 8]);
    *(v8h*)(dst + ((size_t)(b * kC + c0 + c)) * kN + n0 + k8 * 8) = val;
  }
}

// ---------------- projection: dst[16384,64] = src[16384,512] @ wt[512,64] ----------------
__global__ __launch_bounds__(256) void proj_kernel(
    const float* __restrict__ src, const float* __restrict__ wt,
    _Float16* __restrict__ dst) {
  const int lane = threadIdx.x & 31;
  const int w    = threadIdx.x >> 5;
  const int mg = w >> 2, ng = w & 3;
  const int rbase = blockIdx.x * 32 + mg * 16;
  const int nb = ng * 16;
  const int l15 = lane & 15;
  const bool hi = lane >= 16;

  v8f acc = vzero8();
  const float* ap = src + (size_t)(rbase + l15) * kC;

  for (int kb = 0; kb < kC; kb += 32) {
    const int ak = kb + (hi ? 8 : 0);
    float4 a0 = *(const float4*)(ap + ak);
    float4 a1 = *(const float4*)(ap + ak + 4);
    float4 a2 = *(const float4*)(ap + ak + 16);
    float4 a3 = *(const float4*)(ap + ak + 20);
    v16h A;
    A[0]=(_Float16)a0.x; A[1]=(_Float16)a0.y; A[2]=(_Float16)a0.z; A[3]=(_Float16)a0.w;
    A[4]=(_Float16)a1.x; A[5]=(_Float16)a1.y; A[6]=(_Float16)a1.z; A[7]=(_Float16)a1.w;
    A[8]=(_Float16)a2.x; A[9]=(_Float16)a2.y; A[10]=(_Float16)a2.z; A[11]=(_Float16)a2.w;
    A[12]=(_Float16)a3.x; A[13]=(_Float16)a3.y; A[14]=(_Float16)a3.z; A[15]=(_Float16)a3.w;
    const float* bp = wt + (size_t)(kb + (hi ? 8 : 0)) * kD + nb + l15;
    v16h Bf;
#pragma unroll
    for (int i = 0; i < 16; ++i) Bf[i] = (_Float16)bp[((i & 7) + ((i >> 3) << 4)) * kD];
    acc = WMMA16(A, Bf, acc);
  }
#pragma unroll
  for (int v = 0; v < 8; ++v) {
    int row = rbase + v + (hi ? 8 : 0);
    dst[(size_t)row * kD + nb + l15] = (_Float16)acc[v];
  }
}

// ---------------- flash attention + output mix ----------------
// grid: (N/64 q-tiles, 2 channel slabs of 256, B). 8 waves: 4 M-groups x 2 CV-groups.
__global__ __launch_bounds__(256) void attn_kernel(
    const _Float16* __restrict__ q16, const _Float16* __restrict__ k16,
    const _Float16* __restrict__ vT, const float* __restrict__ xin,
    const float* __restrict__ gptr, float* __restrict__ out) {
  __shared__ __attribute__((aligned(16))) _Float16 sVT[2][256 * kVStride]; // [buf][ch][key]
  __shared__ __attribute__((aligned(16))) _Float16 sP[8][16 * kPStride];

  const int tid = threadIdx.x;
  const int lane = tid & 31;
  const int w = tid >> 5;
  const int mg = w >> 1, cg = w & 1;
  const int l15 = lane & 15;
  const bool hi = lane >= 16;

  const int b = blockIdx.z;
  const int qbase = blockIdx.x * 64 + mg * 16;
  const int cvb0 = blockIdx.y * 256;

  // ---- stage one 256ch x 64key V chunk into an LDS buffer ----
#if HAS_TDM
  // Tensor Data Mover: one instruction moves the 32KB tile; D# pad fields
  // reproduce the 72-half padded row stride (32 DW data + 4 DW pad per row).
  auto stage = [&](int kt, int buf) {
    if (w != 0) return;  // TDM is per-wave (EXEC ignored): issue once per block
    const int kk = kt * kKC;
    char* lp = (char*)&sVT[buf][0];
    unsigned lds_off = (unsigned)(size_t)(__attribute__((address_space(3))) char*)lp;
    unsigned long long ga =
        (unsigned long long)(size_t)(vT + ((size_t)(b * kC + cvb0)) * kN + kk);
    uint4v g0;
    g0[0] = 1u;                                  // count=1, user descriptor
    g0[1] = lds_off;                             // lds_addr
    g0[2] = (unsigned)(ga & 0xffffffffu);        // global_addr[31:0]
    g0[3] = (unsigned)((ga >> 32) & 0x01ffffffu) // global_addr[56:32]
            | 0x80000000u;                       // type=2 ("image")
    int8v g1;
    g1[0] = (int)((1u << 16)      // data_size = 2B
                | (1u << 20)      // pad_enable
                | (4u << 22)      // pad_interval: 32 DWORDs (one 64-key row)
                | (3u << 25));    // pad_amount: 4 DWORDs (16B)
    g1[1] = (int)(((unsigned)kN & 0xffffu) << 16);                 // tensor_dim0 lo16
    g1[2] = (int)((((unsigned)kN >> 16) & 0xffffu)
                | (((unsigned)kC & 0xffffu) << 16));               // dim0 hi | dim1 lo
    g1[3] = (int)((((unsigned)kC >> 16) & 0xffffu) | (64u << 16)); // dim1 hi | tile_dim0
    g1[4] = 256;                    // tile_dim1=256, tile_dim2=0 (2D)
    g1[5] = kN;                     // tensor_dim0_stride lo32 = 4096 elements
    g1[6] = 0;
    g1[7] = 0;
    int4v gz = {0, 0, 0, 0};
    int8v gz8 = {0, 0, 0, 0, 0, 0, 0, 0};
    __builtin_amdgcn_tensor_load_to_lds(g0, g1, gz, gz, gz8, 0);
  };
#define STAGE_WAIT() __builtin_amdgcn_s_wait_tensorcnt(0)
#else
  auto stage = [&](int kt, int buf) {
    const int kk = kt * kKC;
#pragma unroll
    for (int it = 0; it < 8; ++it) {
      int flat = it * 256 + tid;           // 2048 x 16B, 128B/row coalesced
      int c    = flat >> 3;
      int k16i = flat & 7;
      const uint4* gsrc =
          (const uint4*)(vT + ((size_t)(b * kC + cvb0 + c)) * kN + kk) + k16i;
      _Float16* ldst = &sVT[buf][c * kVStride + k16i * 8];
#if HAS_ASYNC_LDS
      __builtin_amdgcn_global_load_async_to_lds_b128((int4v*)gsrc, (int4v*)ldst, 0, 0);
#else
      *(uint4*)ldst = *gsrc;
#endif
    }
  };
#define STAGE_WAIT() ASYNC_WAIT()
#endif

  const _Float16* qp = q16 + ((size_t)(b * kN) + qbase + l15) * kD;
  const int ao = hi ? 16 : 0;
  v16h aq0 = cat16(*(const v8h*)(qp + ao),      *(const v8h*)(qp + ao + 8));
  v16h aq1 = cat16(*(const v8h*)(qp + 32 + ao), *(const v8h*)(qp + 32 + ao + 8));

  v8f acc[8];
#pragma unroll
  for (int t = 0; t < 8; ++t) acc[t] = vzero8();
  float mrun[8], lrun[8];
#pragma unroll
  for (int v = 0; v < 8; ++v) { mrun[v] = -__builtin_inff(); lrun[v] = 0.f; }

  stage(0, 0);
  STAGE_WAIT();
  __syncthreads();

  for (int kt = 0; kt < kNT; ++kt) {
    const int kk = kt * kKC;
    const _Float16* sv = sVT[kt & 1];

    // issue next chunk's copy into the other buffer (overlaps the WMMAs)
    if (kt + 1 < kNT) stage(kt + 1, (kt + 1) & 1);
    if (kt + 2 < kNT) {  // warm GL2 one stage further ahead (global_prefetch_b8)
      const _Float16* np = vT + ((size_t)(b * kC + cvb0 + tid)) * kN + kk + 2 * kKC;
      __builtin_prefetch(np, 0, 3);
    }

    // ---- S = Q @ K^T over this chunk ----
    v8f st[4];
#pragma unroll
    for (int j = 0; j < 4; ++j) {
      const _Float16* kp =
          k16 + ((size_t)(b * kN + kk + j * 16 + l15)) * kD + (hi ? 16 : 0);
      v16h bk0 = *(const v16h*)(kp);
      v16h bk1 = *(const v16h*)(kp + 32);
      v8f c = vzero8();
      c = WMMA16(aq0, bk0, c);
      c = WMMA16(aq1, bk1, c);
      st[j] = c;
    }

    // ---- online softmax (rows live across one 16-lane half) ----
    float scl[8];
#pragma unroll
    for (int v = 0; v < 8; ++v) {
      float cm = fmaxf(fmaxf(st[0][v], st[1][v]), fmaxf(st[2][v], st[3][v]));
      cm = fmaxf(cm, __shfl_xor(cm, 1, 32));
      cm = fmaxf(cm, __shfl_xor(cm, 2, 32));
      cm = fmaxf(cm, __shfl_xor(cm, 4, 32));
      cm = fmaxf(cm, __shfl_xor(cm, 8, 32));
      float mn = fmaxf(mrun[v], cm);
      scl[v] = __expf(mrun[v] - mn);
      float rs = 0.f;
#pragma unroll
      for (int j = 0; j < 4; ++j) {
        float p = __expf(st[j][v] - mn);
        st[j][v] = p;
        rs += p;
      }
      rs += __shfl_xor(rs, 1, 32);
      rs += __shfl_xor(rs, 2, 32);
      rs += __shfl_xor(rs, 4, 32);
      rs += __shfl_xor(rs, 8, 32);
      lrun[v] = lrun[v] * scl[v] + rs;
      mrun[v] = mn;
    }
#pragma unroll
    for (int t = 0; t < 8; ++t)
#pragma unroll
      for (int v = 0; v < 8; ++v) acc[t][v] *= scl[v];

    // ---- P: D-layout -> A-fragment re-layout via per-wave LDS ----
    _Float16* sp = sP[w];
#pragma unroll
    for (int j = 0; j < 4; ++j)
#pragma unroll
      for (int v = 0; v < 8; ++v)
        sp[(v + (hi ? 8 : 0)) * kPStride + j * 16 + l15] = (_Float16)st[j][v];
    asm volatile("s_wait_dscnt 0" ::: "memory");
    const _Float16* pr = sp + (size_t)l15 * kPStride;
    v16h pa0 = cat16(*(const v8h*)(pr + ao),      *(const v8h*)(pr + ao + 8));
    v16h pa1 = cat16(*(const v8h*)(pr + 32 + ao), *(const v8h*)(pr + 32 + ao + 8));

    // ---- O += P @ V: contiguous b128 LDS fragment loads ----
#pragma unroll
    for (int t = 0; t < 8; ++t) {
      const _Float16* vp = &sv[(cg * 128 + t * 16 + l15) * kVStride];
      const int ko = hi ? 16 : 0;
      v16h bv0 = cat16(*(const v8h*)(vp + ko),      *(const v8h*)(vp + ko + 8));
      v16h bv1 = cat16(*(const v8h*)(vp + 32 + ko), *(const v8h*)(vp + 32 + ko + 8));
      acc[t] = WMMA16(pa0, bv0, acc[t]);
      acc[t] = WMMA16(pa1, bv1, acc[t]);
    }

    // next buffer resident + everyone done with this one
    STAGE_WAIT();
    __syncthreads();
  }

  // ---- epilogue: out = gamma * (acc / l) + x ----
  const float gamma = *gptr;
#pragma unroll
  for (int v = 0; v < 8; ++v) {
    const int row = qbase + v + (hi ? 8 : 0);
    const float inv = 1.0f / lrun[v];
#pragma unroll
    for (int t = 0; t < 8; ++t) {
      const int ch = cvb0 + cg * 128 + t * 16 + l15;
      const size_t idx = ((size_t)(b * kN) + row) * kC + ch;
      out[idx] = gamma * (acc[t][v] * inv) + xin[idx];
    }
  }
}

extern "C" void kernel_launch(void* const* d_in, const int* in_sizes, int n_in,
                              void* d_out, int out_size, void* d_ws, size_t ws_size,
                              hipStream_t stream) {
  const float* x  = (const float*)d_in[0];
  const float* h  = (const float*)d_in[1];
  const float* f  = (const float*)d_in[2];
  const float* g  = (const float*)d_in[3];
  const float* gm = (const float*)d_in[4];
  float* out = (float*)d_out;
  (void)in_sizes; (void)n_in; (void)out_size; (void)ws_size;

  // workspace: q16 (2MB) | k16 (2MB) | vT (16MB, [b][ch][key] f16)
  _Float16* q16 = (_Float16*)d_ws;
  _Float16* k16 = q16 + (size_t)kB * kN * kD;
  _Float16* vT  = k16 + (size_t)kB * kN * kD;

  transpose_cast_kernel<<<dim3(kN / 64, kC / 64, kB), 256, 0, stream>>>(h, vT);
  proj_kernel<<<dim3((kB * kN) / 32), 256, 0, stream>>>(x, g, q16);  // queries = x @ g
  proj_kernel<<<dim3((kB * kN) / 32), 256, 0, stream>>>(h, f, k16);  // keys    = h @ f
  attn_kernel<<<dim3(kN / 64, 2, kB), 256, 0, stream>>>(q16, k16, vT, x, gm, out);
}
